// WorldModelGRU_72490458022252
// MI455X (gfx1250) — matrix operand order repaired and checked by reading, not compiled, not run on hardware
//
#include <hip/hip_runtime.h>
#include <hip/hip_bf16.h>

// ---------------- types for WMMA fragments ----------------
typedef __attribute__((ext_vector_type(16))) __bf16 v16bf;
typedef __attribute__((ext_vector_type(8)))  float  v8f;
typedef __attribute__((ext_vector_type(4)))  unsigned int u32x4;

union Frag { u32x4 q[2]; v16bf v; };   // 32 bytes: one 16-half bf16 operand

__device__ __forceinline__ unsigned short f2bf(float f) {
  unsigned int u = __float_as_uint(f);
  u += 0x7FFFu + ((u >> 16) & 1u);     // round-to-nearest-even
  return (unsigned short)(u >> 16);
}
__device__ __forceinline__ float sigmoidf_(float x) { return 1.0f / (1.0f + __expf(-x)); }

#define HH   1024
#define BB   2048
#define LDP  1032          // padded LDS row stride (halves) -> avoids 16-way bank conflicts

// ---------------- generic fp32 -> bf16 weight conversion ----------------
__global__ void cvt_bf16_kernel(const float* __restrict__ src,
                                unsigned short* __restrict__ dst, int n) {
  int i = blockIdx.x * blockDim.x + threadIdx.x;
  if (i < n) dst[i] = f2bf(src[i]);
}

// ---------------- action embedding: [B,3] @ [3,1024] + b -> bf16 ----------------
__global__ void action_embed_kernel(const float* __restrict__ action,
                                    const float* __restrict__ W_act,
                                    const float* __restrict__ b_act,
                                    unsigned short* __restrict__ ae) {
  int i = blockIdx.x * 256 + threadIdx.x;         // BB*HH threads
  int b = i >> 10, j = i & 1023;
  float s = b_act[j]
          + action[b * 3 + 0] * W_act[j * 3 + 0]
          + action[b * 3 + 1] * W_act[j * 3 + 1]
          + action[b * 3 + 2] * W_act[j * 3 + 2];
  ae[i] = f2bf(s);
}

// ---------------- token projection: X[t][b][:] = emb[tok[b][t]] @ Wproj^T + b_proj ----------------
__global__ __launch_bounds__(256) void tok_proj_kernel(
    const int* __restrict__ tokens,          // [B,16]
    const float* __restrict__ emb,           // [512,256]
    const unsigned short* __restrict__ Wp,   // bf16 [1024,256] row-major
    const float* __restrict__ b_proj,
    unsigned short* __restrict__ X) {        // bf16 [15][B][1024]
  __shared__ unsigned short lds_e[16 * 264];
  const int t  = blockIdx.x >> 7;            // 0..14
  const int b0 = (blockIdx.x & 127) * 16;
  const int tid = threadIdx.x;
  for (int i = tid; i < 16 * 256; i += 256) {
    int r = i >> 8, c = i & 255;
    int tok = tokens[(b0 + r) * 16 + t];
    lds_e[r * 264 + c] = f2bf(emb[tok * 256 + c]);
  }
  __syncthreads();
  const int lane  = tid & 31;
  const int w     = tid >> 5;
  const int lhalf = lane >> 4;
  const int lmod  = lane & 15;
  const int koffA = 8 * lhalf, koffB = 8 * lhalf;
  const unsigned short* er = &lds_e[lmod * 264];

  for (int tt = 0; tt < 8; ++tt) {
    const int n0 = w * 128 + tt * 16;
    v8f acc = (v8f)0.0f;
    const unsigned short* wr = Wp + (size_t)(n0 + lmod) * 256;
#pragma unroll
    for (int kt = 0; kt < 8; ++kt) {
      const int k0 = kt * 32;
      Frag a, b;
      a.q[0] = *(const u32x4*)(er + k0 + koffA);
      a.q[1] = *(const u32x4*)(er + k0 + koffA + 16);
      b.q[0] = *(const u32x4*)(wr + k0 + koffB);
      b.q[1] = *(const u32x4*)(wr + k0 + koffB + 16);
      acc = __builtin_amdgcn_wmma_f32_16x16x32_bf16(false, a.v, false, b.v,
                                                    (short)0, acc, false, false);
    }
    const int col = n0 + lmod;
    const float bp = b_proj[col];
#pragma unroll
    for (int v = 0; v < 8; ++v) {
      const int m = v + 8 * lhalf;
      X[((size_t)t * BB + b0 + m) * HH + col] = f2bf(acc[v] + bp);
    }
  }
}

// ---------------- fused GRU step (+ optional logits head) ----------------
// grid 128, block 256 (8 wave32). Each block: 16 batch rows.
// Each wave owns matched 16-col slices of r/z/n so gates live in registers.
template <bool HAS_X, bool HAS_HEAD>
__global__ __launch_bounds__(256) void gru_step_kernel(
    const float* __restrict__ h_in,            // [B,H] f32
    const unsigned short* __restrict__ x_bf,   // [B,H] bf16 (or null)
    const unsigned short* __restrict__ Whh,    // bf16 [3H,H]
    const unsigned short* __restrict__ Wih,    // bf16 [3H,H]
    const float* __restrict__ b_ih,
    const float* __restrict__ b_hh,
    float* __restrict__ h_out,                 // [B,H] f32
    const unsigned short* __restrict__ Whead,  // bf16 [512,H] (or null)
    const float* __restrict__ b_head,
    float* __restrict__ logits) {              // d_out + s*512, row stride 8192 (or null)
  __shared__ unsigned short lds_h [16 * LDP];  // h  bf16 (GEMM A operand)
  __shared__ unsigned short lds_x [16 * LDP];  // x  bf16 (GEMM A operand)
  __shared__ unsigned short lds_hn[16 * LDP];  // h' bf16 (head A operand)
  const int b0  = blockIdx.x * 16;
  const int tid = threadIdx.x;

  for (int i = tid; i < 16 * HH; i += 256) {   // stage h -> bf16 LDS (+ x copy)
    int r = i >> 10, c = i & 1023;
    lds_h[r * LDP + c] = f2bf(h_in[(size_t)(b0 + r) * HH + c]);
    if (HAS_X) lds_x[r * LDP + c] = x_bf[(size_t)(b0 + r) * HH + c];
  }
  __syncthreads();

  const int lane  = tid & 31;
  const int w     = tid >> 5;
  const int lhalf = lane >> 4;
  const int lmod  = lane & 15;
  const int koffA = 8 * lhalf, koffB = 8 * lhalf;
  const unsigned short* hr = &lds_h[lmod * LDP];
  const unsigned short* xr = &lds_x[lmod * LDP];

  for (int t = 0; t < 8; ++t) {
    const int col = w * 128 + t * 16;          // column base within H
    v8f ar = (v8f)0.0f, az = (v8f)0.0f;        // r,z: x-path + h-path combined
    v8f ani = (v8f)0.0f, anh = (v8f)0.0f;      // n-gate: keep paths separate
    const unsigned short* whr = Whh + (size_t)(col + lmod) * HH;
    const unsigned short* wir = Wih + (size_t)(col + lmod) * HH;
#pragma unroll 2
    for (int kt = 0; kt < 32; ++kt) {
      const int k0 = kt * 32;
      Frag ah, br, bz, bn;
      ah.q[0] = *(const u32x4*)(hr + k0 + koffA);
      ah.q[1] = *(const u32x4*)(hr + k0 + koffA + 16);
      const unsigned short* p = whr + k0 + koffB;
      br.q[0] = *(const u32x4*)(p);                 br.q[1] = *(const u32x4*)(p + 16);
      bz.q[0] = *(const u32x4*)(p + 1048576);       bz.q[1] = *(const u32x4*)(p + 1048576 + 16);
      bn.q[0] = *(const u32x4*)(p + 2097152);       bn.q[1] = *(const u32x4*)(p + 2097152 + 16);
      ar  = __builtin_amdgcn_wmma_f32_16x16x32_bf16(false, ah.v, false, br.v, (short)0, ar,  false, false);
      az  = __builtin_amdgcn_wmma_f32_16x16x32_bf16(false, ah.v, false, bz.v, (short)0, az,  false, false);
      anh = __builtin_amdgcn_wmma_f32_16x16x32_bf16(false, ah.v, false, bn.v, (short)0, anh, false, false);
      if (HAS_X) {
        Frag ax, cr, cz, cn;
        ax.q[0] = *(const u32x4*)(xr + k0 + koffA);
        ax.q[1] = *(const u32x4*)(xr + k0 + koffA + 16);
        const unsigned short* q2 = wir + k0 + koffB;
        cr.q[0] = *(const u32x4*)(q2);               cr.q[1] = *(const u32x4*)(q2 + 16);
        cz.q[0] = *(const u32x4*)(q2 + 1048576);     cz.q[1] = *(const u32x4*)(q2 + 1048576 + 16);
        cn.q[0] = *(const u32x4*)(q2 + 2097152);     cn.q[1] = *(const u32x4*)(q2 + 2097152 + 16);
        ar  = __builtin_amdgcn_wmma_f32_16x16x32_bf16(false, ax.v, false, cr.v, (short)0, ar,  false, false);
        az  = __builtin_amdgcn_wmma_f32_16x16x32_bf16(false, ax.v, false, cz.v, (short)0, az,  false, false);
        ani = __builtin_amdgcn_wmma_f32_16x16x32_bf16(false, ax.v, false, cn.v, (short)0, ani, false, false);
      }
    }
    // gates in registers
    const int gcol = col + lmod;
    const float bir = b_ih[gcol],        bhr = b_hh[gcol];
    const float biz = b_ih[HH + gcol],   bhz = b_hh[HH + gcol];
    const float bin = b_ih[2*HH + gcol], bhn = b_hh[2*HH + gcol];
#pragma unroll
    for (int v = 0; v < 8; ++v) {
      const int m = v + 8 * lhalf;
      const size_t grow = (size_t)(b0 + m) * HH + gcol;
      float r  = sigmoidf_(ar[v] + bir + bhr);
      float z  = sigmoidf_(az[v] + biz + bhz);
      float n  = tanhf(ani[v] + bin + r * (anh[v] + bhn));
      float hp = h_in[grow];
      float hn = (1.0f - z) * n + z * hp;
      h_out[grow] = hn;
      lds_hn[m * LDP + gcol] = f2bf(hn);
    }
  }
  __syncthreads();

  if (HAS_HEAD) {                                   // head GEMM: h' @ W_head^T + b
    const unsigned short* hnr = &lds_hn[lmod * LDP];
    for (int t2 = 0; t2 < 4; ++t2) {
      const int n0 = w * 64 + t2 * 16;
      v8f c = (v8f)0.0f;
      const unsigned short* wh = Whead + (size_t)(n0 + lmod) * HH;
#pragma unroll 2
      for (int kt = 0; kt < 32; ++kt) {
        const int k0 = kt * 32;
        Frag a, b;
        a.q[0] = *(const u32x4*)(hnr + k0 + koffA);
        a.q[1] = *(const u32x4*)(hnr + k0 + koffA + 16);
        b.q[0] = *(const u32x4*)(wh + k0 + koffB);
        b.q[1] = *(const u32x4*)(wh + k0 + koffB + 16);
        c = __builtin_amdgcn_wmma_f32_16x16x32_bf16(false, a.v, false, b.v,
                                                    (short)0, c, false, false);
      }
      const int ocol = n0 + lmod;
      const float bb = b_head[ocol];
#pragma unroll
      for (int v = 0; v < 8; ++v) {
        const int m = v + 8 * lhalf;
        logits[(size_t)(b0 + m) * 8192 + ocol] = c[v] + bb;
      }
    }
  }
}

// ---------------- reward / done heads (tiny dot products) ----------------
__global__ __launch_bounds__(256) void heads_kernel(
    const float* __restrict__ h, const float* __restrict__ Wr, const float* __restrict__ br,
    const float* __restrict__ Wd, const float* __restrict__ bd,
    float* __restrict__ out_r, float* __restrict__ out_d) {
  __shared__ float sr[256], sd[256];
  const int b = blockIdx.x, tid = threadIdx.x;
  float r = 0.f, d = 0.f;
  for (int k = tid; k < HH; k += 256) {
    float hv = h[(size_t)b * HH + k];
    r += hv * Wr[k]; d += hv * Wd[k];
  }
  sr[tid] = r; sd[tid] = d; __syncthreads();
  for (int s = 128; s > 0; s >>= 1) {
    if (tid < s) { sr[tid] += sr[tid + s]; sd[tid] += sd[tid + s]; }
    __syncthreads();
  }
  if (tid == 0) { out_r[b] = sr[0] + br[0]; out_d[b] = sd[0] + bd[0]; }
}

// ---------------- host orchestration ----------------
extern "C" void kernel_launch(void* const* d_in, const int* in_sizes, int n_in,
                              void* d_out, int out_size, void* d_ws, size_t ws_size,
                              hipStream_t stream) {
  (void)in_sizes; (void)n_in; (void)out_size; (void)ws_size;
  const float* action = (const float*)d_in[0];
  const float* prev_h = (const float*)d_in[1];
  const int*   tokens = (const int*)  d_in[2];
  const float* emb    = (const float*)d_in[3];
  const float* W_act  = (const float*)d_in[4];
  const float* b_act  = (const float*)d_in[5];
  const float* W_proj = (const float*)d_in[6];
  const float* b_proj = (const float*)d_in[7];
  const float* W_ih   = (const float*)d_in[8];
  const float* W_hh   = (const float*)d_in[9];
  const float* b_ih   = (const float*)d_in[10];
  const float* b_hh   = (const float*)d_in[11];
  const float* W_head = (const float*)d_in[12];
  const float* b_head = (const float*)d_in[13];
  const float* W_r    = (const float*)d_in[14];
  const float* b_r    = (const float*)d_in[15];
  const float* W_d    = (const float*)d_in[16];
  const float* b_d    = (const float*)d_in[17];

  float* out        = (float*)d_out;
  float* out_logits = out;                               // [2048][16][512]
  float* out_reward = out + (size_t)BB * 16 * 512;
  float* out_done   = out_reward + BB;
  float* out_finalh = out_done + BB;                     // [2048][1024]

  char* ws = (char*)d_ws; size_t off = 0;
  auto carve = [&](size_t bytes) { void* p = ws + off; off = (off + bytes + 255) & ~(size_t)255; return p; };
  unsigned short* Whh_bf   = (unsigned short*)carve((size_t)3072 * HH * 2);
  unsigned short* Wih_bf   = (unsigned short*)carve((size_t)3072 * HH * 2);
  unsigned short* Whead_bf = (unsigned short*)carve((size_t)512 * HH * 2);
  unsigned short* Wproj_bf = (unsigned short*)carve((size_t)1024 * 256 * 2);
  unsigned short* ae_bf    = (unsigned short*)carve((size_t)BB * HH * 2);
  unsigned short* X_bf     = (unsigned short*)carve((size_t)15 * BB * HH * 2);
  float* h0 = (float*)carve((size_t)BB * HH * 4);
  float* h1 = (float*)carve((size_t)BB * HH * 4);

  // one-time conversions + off-critical-path precompute
  cvt_bf16_kernel<<<(3072 * HH) / 256, 256, 0, stream>>>(W_hh,  Whh_bf,  3072 * HH);
  cvt_bf16_kernel<<<(3072 * HH) / 256, 256, 0, stream>>>(W_ih,  Wih_bf,  3072 * HH);
  cvt_bf16_kernel<<<(512  * HH) / 256, 256, 0, stream>>>(W_head, Whead_bf, 512 * HH);
  cvt_bf16_kernel<<<(1024 * 256) / 256, 256, 0, stream>>>(W_proj, Wproj_bf, 1024 * 256);
  action_embed_kernel<<<(BB * HH) / 256, 256, 0, stream>>>(action, W_act, b_act, ae_bf);
  tok_proj_kernel<<<15 * 128, 256, 0, stream>>>(tokens, emb, Wproj_bf, b_proj, X_bf);

  // transition step (x = action embedding, no logits head)
  gru_step_kernel<true, false><<<128, 256, 0, stream>>>(
      prev_h, ae_bf, Whh_bf, Wih_bf, b_ih, b_hh, h0, nullptr, nullptr, nullptr);
  heads_kernel<<<BB, 256, 0, stream>>>(h0, W_r, b_r, W_d, b_d, out_reward, out_done);

  // teacher-forced scan: 16 dependent steps
  float* cur = h0; float* nxt = h1;
  for (int s = 0; s < 16; ++s) {
    float* hout = (s == 15) ? out_finalh : nxt;
    if (s == 0) {
      gru_step_kernel<false, true><<<128, 256, 0, stream>>>(
          cur, nullptr, Whh_bf, Wih_bf, b_ih, b_hh, hout,
          Whead_bf, b_head, out_logits + (size_t)s * 512);
    } else {
      gru_step_kernel<true, true><<<128, 256, 0, stream>>>(
          cur, X_bf + (size_t)(s - 1) * BB * HH, Whh_bf, Wih_bf, b_ih, b_hh, hout,
          Whead_bf, b_head, out_logits + (size_t)s * 512);
    }
    float* tmp = cur; cur = hout; nxt = tmp;
  }
}
